// MultiHeadAttention_17454747091104
// MI455X (gfx1250) — matrix operand order repaired and checked by reading, not compiled, not run on hardware
//
#include <hip/hip_runtime.h>
#include <hip/hip_bf16.h>

// ---------------- problem constants ----------------
constexpr int B  = 64;
constexpr int S  = 197;   // 196 patches + CLS
constexpr int D  = 768;
constexpr int H  = 12;
constexpr int DH = 64;    // head dim
constexpr int SP = 224;   // S padded to multiple of 32 (14 tiles of 16)
constexpr int NT = SP / 16; // 14 column tiles of the score matrix
constexpr int PMAP = 196; // CLS->patch map entries per (b,h) = 14*14

typedef __attribute__((ext_vector_type(16))) __bf16 v16bf;
typedef __attribute__((ext_vector_type(8)))  __bf16 v8bf;
typedef __attribute__((ext_vector_type(8)))  float  v8f;

// ---------------- WMMA helpers --------------------------------------------
// D = A(16x32 bf16) * B(32x16 bf16) + C(16x16 f32), wave32.
__device__ __forceinline__ v8f wmma_bf16(v16bf a, v16bf b, v8f c) {
  return __builtin_amdgcn_wmma_f32_16x16x32_bf16(
      /*neg_a=*/false, a, /*neg_b=*/false, b,
      /*c_mod=*/(short)0, c, /*reuse_a=*/false, /*reuse_b=*/false);
}

// A fragment, 16-bit 16x32 (MxK) layout per ISA 7.12.2:
//  lane L: M = L%16, g = L/16; element e<8 -> K = k0 + g*8 + e
//                              element e>=8 -> K = k0 + 16 + g*8 + (e-8)
// => two contiguous 8-element (16B) runs per lane.
__device__ __forceinline__ v16bf load_a_bf16(const __bf16* __restrict__ A,
                                             int lda, int m, int g, int k0) {
  const __bf16* p0 = A + (size_t)m * lda + k0 + g * 8;
  v8bf lo = *(const v8bf*)p0;
  v8bf hi = *(const v8bf*)(p0 + 16);
  v16bf f;
#pragma unroll
  for (int i = 0; i < 8; ++i) { f[i] = lo[i]; f[i + 8] = hi[i]; }
  return f;
}

// Same fragment, source fp32 (convert on load). Caller guarantees m in-range.
__device__ __forceinline__ v16bf load_a_f32(const float* __restrict__ A,
                                            int lda, int m, int g, int k0) {
  const float* p0 = A + (size_t)m * lda + k0 + g * 8;
  v8f lo = *(const v8f*)p0;
  v8f hi = *(const v8f*)(p0 + 16);
  v16bf f;
#pragma unroll
  for (int i = 0; i < 8; ++i) { f[i] = (__bf16)lo[i]; f[i + 8] = (__bf16)hi[i]; }
  return f;
}

// B fragment, 16-bit 32x16 (KxN). Bt is stored [N,K] row-major so lane
// (N = n, group g) reads K = k0 + g*8 + e (e<8), K = k0 + 16 + g*8 + (e-8) (e>=8).
__device__ __forceinline__ v16bf load_b_bf16(const __bf16* __restrict__ Bt,
                                             int ldb, int n, int g, int k0) {
  const __bf16* p0 = Bt + (size_t)n * ldb + k0 + g * 8;
  v8bf lo = *(const v8bf*)p0;
  v8bf hi = *(const v8bf*)(p0 + 16);
  v16bf f;
#pragma unroll
  for (int i = 0; i < 8; ++i) { f[i] = lo[i]; f[i + 8] = hi[i]; }
  return f;
}

// ---------------- K1: weights fp32 -> bf16 (layout [out,in] kept: = [N,K]) --
__global__ void wcvt_kernel(const float* __restrict__ Wq, const float* __restrict__ Wk,
                            const float* __restrict__ Wv, const float* __restrict__ Wo,
                            __bf16* __restrict__ dst) {
  const int WN = D * D;
  int i = blockIdx.x * blockDim.x + threadIdx.x;
  if (i >= 4 * WN) return;
  int which = i / WN, j = i - which * WN;
  const float* src = (which == 0) ? Wq : (which == 1) ? Wk : (which == 2) ? Wv : Wo;
  dst[i] = (__bf16)src[j];
}

// ---------------- K2: projection Y = X @ W^T ------------------------------
// grid (SP/16, 3, B), block 128 (4 waves). Wave w owns N-strip (y*4+w)*64.
// mode 0: store [B,H,SP,DH] (qh / kh); mode 2: store transposed vhT [B,H,DH,SP].
// Padded rows (s >= S) are stored as exact zeros (A row clamped for the load,
// zero substituted at store time). k-loop is software-pipelined: fragments for
// k+32 are fetched before the WMMAs for k are issued.
__global__ void proj_kernel(const float* __restrict__ X, const __bf16* __restrict__ Wt,
                            __bf16* __restrict__ out, int mode) {
  int lane = threadIdx.x & 31, wave = threadIdx.x >> 5;
  int ln = lane & 15, g = lane >> 4;
  int mbase = blockIdx.x * 16;
  int nbase = (blockIdx.y * 4 + wave) * 64;
  int b = blockIdx.z;
  int mrow = min(mbase + ln, S - 1);        // always-valid A row
  const float* Xb = X + (size_t)b * S * D;

  v8f acc[4] = {};
  // prologue: fragments for k = 0
  v16bf a = load_a_f32(Xb, D, mrow, g, 0);
  v16bf bb[4];
#pragma unroll
  for (int t = 0; t < 4; ++t) bb[t] = load_b_bf16(Wt, D, nbase + t * 16 + ln, g, 0);

  for (int k = 0; k < D; k += 32) {
    int kn = (k + 32 < D) ? (k + 32) : 0;   // wrapped -> always-valid load
    __builtin_prefetch(Xb + (size_t)mrow * D + k + 128, 0, 3);
    v16bf an = load_a_f32(Xb, D, mrow, g, kn);
    v16bf bn[4];
#pragma unroll
    for (int t = 0; t < 4; ++t)
      bn[t] = load_b_bf16(Wt, D, nbase + t * 16 + ln, g, kn);
#pragma unroll
    for (int t = 0; t < 4; ++t)
      acc[t] = wmma_bf16(a, bb[t], acc[t]);
    a = an;
#pragma unroll
    for (int t = 0; t < 4; ++t) bb[t] = bn[t];
  }

  if (mode != 2) {                          // qh / kh: [B,H,SP,DH]
#pragma unroll
    for (int t = 0; t < 4; ++t) {
      int e = nbase + t * 16 + ln;
      int h = e >> 6, dh = e & 63;
      __bf16* op = out + (((size_t)b * H + h) * SP) * DH + dh;
#pragma unroll
      for (int r = 0; r < 8; ++r) {
        int s = mbase + g * 8 + r;          // C layout: VGPR r -> M = g*8+r
        float vv = (s < S) ? acc[t][r] : 0.0f;
        op[(size_t)s * DH] = (__bf16)vv;
      }
    }
  } else {                                  // vhT: [B,H,DH,SP], contiguous in s
#pragma unroll
    for (int t = 0; t < 4; ++t) {
      int e = nbase + t * 16 + ln;
      int h = e >> 6, dh = e & 63;
      int s0 = mbase + g * 8;
      v8bf ov;
#pragma unroll
      for (int r = 0; r < 8; ++r) {
        float vv = (s0 + r < S) ? acc[t][r] : 0.0f;
        ov[r] = (__bf16)vv;
      }
      *(v8bf*)(out + (((size_t)b * H + h) * DH + dh) * SP + s0) = ov;
    }
  }
}

// ---------------- K3: fused scores -> softmax -> CLS map -> P @ V ---------
// grid (13, B*H), block 32 (one wave). Row tile = 16 rows x 224 cols.
__global__ void attn_av_kernel(const __bf16* __restrict__ qh, const __bf16* __restrict__ kh,
                               const __bf16* __restrict__ vhT, __bf16* __restrict__ x,
                               float* __restrict__ clsmap) {
  __shared__ __align__(16) __bf16 pls[16 * SP];  // 7 KB prob tile, C->A reshape
  int lane = threadIdx.x & 31;
  int ln = lane & 15, g = lane >> 4;
  int mbase = blockIdx.x * 16;
  int bh = blockIdx.y;
  int b = bh / H, h = bh % H;
  const __bf16* Q  = qh  + (size_t)bh * SP * DH;
  const __bf16* K  = kh  + (size_t)bh * SP * DH;
  const __bf16* Vt = vhT + (size_t)bh * DH * SP;

  // ---- scores = (Q @ K^T) / 8 : 14 N-tiles, K-loop over DH=64 ----
  v8f acc[NT] = {};
#pragma unroll
  for (int k = 0; k < DH; k += 32) {
    v16bf a = load_a_bf16(Q, DH, mbase + ln, g, k);
#pragma unroll
    for (int tt = 0; tt < NT; tt += 7) {    // batches of 7 -> load clause + MLP
      v16bf bbuf[7];
#pragma unroll
      for (int j = 0; j < 7; ++j)
        bbuf[j] = load_b_bf16(K, DH, (tt + j) * 16 + ln, g, k);
#pragma unroll
      for (int j = 0; j < 7; ++j)
        acc[tt + j] = wmma_bf16(a, bbuf[j], acc[tt + j]);
    }
  }

  // ---- softmax over 197 real columns (cols >= S masked) ----
  const float scale = 0.125f;               // 1/sqrt(64)
  float rmax[8], rsum[8];
#pragma unroll
  for (int r = 0; r < 8; ++r) rmax[r] = -3.4e38f;
#pragma unroll
  for (int t = 0; t < NT; ++t) {
    int col = t * 16 + ln;
    bool cv = col < S;
#pragma unroll
    for (int r = 0; r < 8; ++r) {
      float vv = acc[t][r] * scale;
      vv = cv ? vv : -3.4e38f;
      acc[t][r] = vv;
      rmax[r] = fmaxf(rmax[r], vv);
    }
  }
#pragma unroll
  for (int r = 0; r < 8; ++r)               // row lives in one 16-lane group
    for (int off = 1; off < 16; off <<= 1)
      rmax[r] = fmaxf(rmax[r], __shfl_xor(rmax[r], off, 32));
#pragma unroll
  for (int r = 0; r < 8; ++r) rsum[r] = 0.0f;
#pragma unroll
  for (int t = 0; t < NT; ++t) {
    int col = t * 16 + ln;
    bool cv = col < S;
#pragma unroll
    for (int r = 0; r < 8; ++r) {
      float p = cv ? __expf(acc[t][r] - rmax[r]) : 0.0f;
      acc[t][r] = p;
      rsum[r] += p;
    }
  }
#pragma unroll
  for (int r = 0; r < 8; ++r) {
    for (int off = 1; off < 16; off <<= 1) rsum[r] += __shfl_xor(rsum[r], off, 32);
    rsum[r] = 1.0f / rsum[r];
  }

  // ---- write prob tile to LDS (and CLS->patch map from row 0) ----
#pragma unroll
  for (int t = 0; t < NT; ++t) {
    int col = t * 16 + ln;
#pragma unroll
    for (int r = 0; r < 8; ++r) {
      float p = acc[t][r] * rsum[r];
      pls[(g * 8 + r) * SP + col] = (__bf16)p;
      if (blockIdx.x == 0 && g == 0 && r == 0 && col >= 1 && col < S)
        clsmap[(size_t)bh * PMAP + (col - 1)] = p;  // [B,H,14,14] flat
    }
  }
  __syncthreads();

  // ---- x_tile = P @ V : A from LDS, B = vhT [N=dh, K=s], pipelined ----
  v8f acc2[4] = {};
  v16bf vb[4];
#pragma unroll
  for (int t = 0; t < 4; ++t) vb[t] = load_b_bf16(Vt, SP, t * 16 + ln, g, 0);
#pragma unroll
  for (int k = 0; k < SP; k += 32) {
    int kn = (k + 32 < SP) ? (k + 32) : 0;
    v16bf vn[4];
#pragma unroll
    for (int t = 0; t < 4; ++t)
      vn[t] = load_b_bf16(Vt, SP, t * 16 + ln, g, kn);
    const __bf16* p0 = &pls[ln * SP + k + g * 8];
    v8bf lo = *(const v8bf*)p0;
    v8bf hi = *(const v8bf*)(p0 + 16);
    v16bf a;
#pragma unroll
    for (int i = 0; i < 8; ++i) { a[i] = lo[i]; a[i + 8] = hi[i]; }
#pragma unroll
    for (int t = 0; t < 4; ++t)
      acc2[t] = wmma_bf16(a, vb[t], acc2[t]);
#pragma unroll
    for (int t = 0; t < 4; ++t) vb[t] = vn[t];
  }
  // store x [B,SP,D] at column h*64 + n (row-major, K-major for final GEMM)
#pragma unroll
  for (int t = 0; t < 4; ++t) {
    int dcol = h * 64 + t * 16 + ln;
#pragma unroll
    for (int r = 0; r < 8; ++r) {
      int s = mbase + g * 8 + r;
      x[((size_t)b * SP + s) * D + dcol] = (__bf16)acc2[t][r];
    }
  }
}

// ---------------- K4: output projection out = x @ Wo^T (fp32 out) ---------
// grid (13, 3, B), block 128 (4 waves). Software-pipelined like proj_kernel.
__global__ void oproj_kernel(const __bf16* __restrict__ x, const __bf16* __restrict__ WoT,
                             float* __restrict__ out) {
  int lane = threadIdx.x & 31, wave = threadIdx.x >> 5;
  int ln = lane & 15, g = lane >> 4;
  int mbase = blockIdx.x * 16;
  int nbase = (blockIdx.y * 4 + wave) * 64;
  int b = blockIdx.z;
  const __bf16* Xb = x + (size_t)b * SP * D;

  v8f acc[4] = {};
  v16bf a = load_a_bf16(Xb, D, mbase + ln, g, 0);
  v16bf bb[4];
#pragma unroll
  for (int t = 0; t < 4; ++t) bb[t] = load_b_bf16(WoT, D, nbase + t * 16 + ln, g, 0);

  for (int k = 0; k < D; k += 32) {
    int kn = (k + 32 < D) ? (k + 32) : 0;
    __builtin_prefetch(Xb + (size_t)(mbase + ln) * D + k + 128, 0, 3);
    v16bf an = load_a_bf16(Xb, D, mbase + ln, g, kn);
    v16bf bn[4];
#pragma unroll
    for (int t = 0; t < 4; ++t)
      bn[t] = load_b_bf16(WoT, D, nbase + t * 16 + ln, g, kn);
#pragma unroll
    for (int t = 0; t < 4; ++t)
      acc[t] = wmma_bf16(a, bb[t], acc[t]);
    a = an;
#pragma unroll
    for (int t = 0; t < 4; ++t) bb[t] = bn[t];
  }
#pragma unroll
  for (int t = 0; t < 4; ++t) {
    int e = nbase + t * 16 + ln;
#pragma unroll
    for (int r = 0; r < 8; ++r) {
      int s = mbase + g * 8 + r;
      if (s < S) out[((size_t)b * S + s) * D + e] = acc[t][r];
    }
  }
}

// ---------------- host-side launcher --------------------------------------
extern "C" void kernel_launch(void* const* d_in, const int* in_sizes, int n_in,
                              void* d_out, int out_size, void* d_ws, size_t ws_size,
                              hipStream_t stream) {
  (void)in_sizes; (void)n_in; (void)out_size; (void)ws_size;
  const float* q  = (const float*)d_in[0];
  const float* k  = (const float*)d_in[1];
  const float* v  = (const float*)d_in[2];
  const float* Wq = (const float*)d_in[3];
  const float* Wk = (const float*)d_in[4];
  const float* Wv = (const float*)d_in[5];
  const float* Wo = (const float*)d_in[6];

  // workspace layout (bf16): [Wq|Wk|Wv|Wo][qh][kh][vhT][x]  ~= 93 MB
  const size_t WN = (size_t)D * D;
  const size_t HS = (size_t)B * H * SP * DH;
  __bf16* wbf = (__bf16*)d_ws;
  __bf16* qh  = wbf + 4 * WN;
  __bf16* kh  = qh + HS;
  __bf16* vhT = kh + HS;
  __bf16* xws = vhT + HS;

  float* out = (float*)d_out;
  float* clsmap = out + (size_t)B * S * D;   // second tuple output [B,H,14,14]

  wcvt_kernel<<<dim3((unsigned)((4 * WN + 255) / 256)), 256, 0, stream>>>(Wq, Wk, Wv, Wo, wbf);

  dim3 gp(SP / 16, 3, B);
  proj_kernel<<<gp, 128, 0, stream>>>(q, wbf,          qh,  0);
  proj_kernel<<<gp, 128, 0, stream>>>(k, wbf + WN,     kh,  0);
  proj_kernel<<<gp, 128, 0, stream>>>(v, wbf + 2 * WN, vhT, 2);

  attn_av_kernel<<<dim3(13, B * H), 32, 0, stream>>>(qh, kh, vhT, xws, clsmap);

  oproj_kernel<<<dim3(13, 3, B), 128, 0, stream>>>(xws, wbf + 3 * WN, out);
}
